// CategoricalGraphAtt_27522150432930
// MI455X (gfx1250) — matrix operand order repaired and checked by reading, not compiled, not run on hardware
//
#include <hip/hip_runtime.h>
#include <math.h>

// ---------------------------------------------------------------------------
// CategoricalGraphAtt on MI455X (gfx1250): bf16 WMMA GEMMs + fp32 accumulate.
// All GEMM fragment loads are contiguous global_load_b128 (seq is transposed
// once so the attention GEMM also uses the N x K "Bt" operand form).
// ---------------------------------------------------------------------------

typedef __attribute__((ext_vector_type(16))) __bf16 v16bf;
typedef __attribute__((ext_vector_type(8)))  float  v8f;

#define T_DIM  512
#define IN_D   256
#define H_DIM  1024
#define NNODES 100
#define NPAD   112   // 7 * 16, padded row count for the GRU state
#define NCAT   5
#define NPER   20

union FragU { uint4 u[2]; v16bf v; };

// A-matrix 16x32 bf16 fragment (row-major A, lda in elements), no row guard.
// lanes 0-15: row=lane,    K = {0..7, 16..23} (+kbase)
// lanes16-31: row=lane-16, K = {8..15,24..31} (+kbase)
__device__ __forceinline__ v16bf load_a_frag_nc(const __bf16* A, long lda,
                                                int mbase, int kbase) {
  int lane = threadIdx.x & 31;
  int row  = mbase + (lane & 15);
  int k0   = kbase + ((lane & 16) ? 8 : 0);
  FragU f;
  const __bf16* p = A + (long)row * lda + k0;
  f.u[0] = *(const uint4*)(p);
  f.u[1] = *(const uint4*)(p + 16);
  return f.v;
}

// Guarded variant: out-of-range rows contribute zeros.
__device__ __forceinline__ v16bf load_a_frag_g(const __bf16* A, long lda,
                                               int mbase, int kbase, int M) {
  int lane = threadIdx.x & 31;
  int row  = mbase + (lane & 15);
  int k0   = kbase + ((lane & 16) ? 8 : 0);
  FragU f;
  if (row < M) {
    const __bf16* p = A + (long)row * lda + k0;
    f.u[0] = *(const uint4*)(p);
    f.u[1] = *(const uint4*)(p + 16);
  } else {
    f.u[0] = make_uint4(0u, 0u, 0u, 0u);
    f.u[1] = make_uint4(0u, 0u, 0u, 0u);
  }
  return f.v;
}

// B-matrix 32x16 fragment from a transposed operand (Bt is N x K row-major,
// the "x @ W.T" pattern). lane: col N = nbase + lane%16, K = {0..7,16..23} (lanes 0-15) / {8..15,24..31} (lanes 16-31).
__device__ __forceinline__ v16bf load_bt_frag(const __bf16* Bt, long ldb,
                                              int nbase, int kbase) {
  int lane = threadIdx.x & 31;
  int ncol = nbase + (lane & 15);
  int k0   = kbase + ((lane & 16) ? 8 : 0);
  FragU f;
  const __bf16* p = Bt + (long)ncol * ldb + k0;
  f.u[0] = *(const uint4*)(p);
  f.u[1] = *(const uint4*)(p + 16);
  return f.v;
}

__device__ __forceinline__ v8f wmma_bf16(v16bf a, v16bf b, v8f c) {
  return __builtin_amdgcn_wmma_f32_16x16x32_bf16(false, a, false, b,
                                                 (short)0, c, false, false);
}

// ---------------------------------------------------------------------------
// Generic tiled GEMM: C[M,N] = A[M,K] (bf16) @ Bt[N,K]^T (bf16) + bias.
// One wave per 16x16 C tile. GUARD=false requires M % 16 == 0.
// ---------------------------------------------------------------------------
template <bool RELU, bool GUARD, typename CT>
__global__ void wmma_gemm(const __bf16* __restrict__ A, long lda, long sA,
                          const __bf16* __restrict__ Bt, long ldb, long sB,
                          CT* __restrict__ C, long ldc, long sC,
                          int M, int N, int K, const float* __restrict__ bias) {
  int wave   = (int)((blockIdx.x * blockDim.x + threadIdx.x) >> 5);
  int ntiles = N >> 4;
  int mtiles = (M + 15) >> 4;
  if (wave >= ntiles * mtiles) return;
  int mt = wave / ntiles, nt = wave % ntiles;

  const __bf16* Ab = A  + (long)blockIdx.y * sA;
  const __bf16* Bb = Bt + (long)blockIdx.y * sB;
  CT*           Cb = C  + (long)blockIdx.y * sC;

  v8f acc = {};
  for (int k = 0; k < K; k += 32) {
    v16bf a = GUARD ? load_a_frag_g(Ab, lda, mt * 16, k, M)
                    : load_a_frag_nc(Ab, lda, mt * 16, k);
    v16bf b = load_bt_frag(Bb, ldb, nt * 16, k);
    acc = wmma_bf16(a, b, acc);
  }

  int lane  = threadIdx.x & 31;
  int ncol  = nt * 16 + (lane & 15);
  int rbase = mt * 16 + ((lane & 16) ? 8 : 0);
  float badd = bias ? bias[ncol] : 0.0f;
#pragma unroll
  for (int e = 0; e < 8; ++e) {
    int row = rbase + e;
    if (!GUARD || row < M) {
      float x = acc[e] + badd;
      if (RELU) x = fmaxf(x, 0.0f);
      Cb[(long)row * ldc + ncol] = (CT)x;
    }
  }
}

// ---------------------------------------------------------------------------
// One GRU time step. gh = h_{t-1} @ Whh^T for the 3 gates via WMMA, then the
// gate nonlinearity directly on the accumulator layout. Launched 512x.
// seqBf is padded to NPAD rows; pad rows are kept at zero each step so the
// A-fragment loads need no predication.
// seqBf layout: [NPAD][T][H] bf16; seqF: [NNODES][T][H] f32;
// xp (bf16): [NNODES][T][3H] (includes bih).
// ---------------------------------------------------------------------------
__global__ void gru_step(__bf16* __restrict__ seqBf, float* __restrict__ seqF,
                         const __bf16* __restrict__ Whh,
                         const __bf16* __restrict__ xp,
                         const float* __restrict__ bhh, int t) {
  const int ntiles = H_DIM >> 4;  // 64
  const int mtiles = NPAD >> 4;   // 7
  int wave = (int)((blockIdx.x * blockDim.x + threadIdx.x) >> 5);
  if (wave >= ntiles * mtiles) return;
  int mt = wave / ntiles, nt = wave % ntiles;

  v8f accr = {}, accz = {}, accn = {};
  if (t > 0) {
    const __bf16* Aprev = seqBf + (long)(t - 1) * H_DIM;  // row stride T*H
    const long lda = (long)T_DIM * H_DIM;
    for (int k = 0; k < H_DIM; k += 32) {
      v16bf a  = load_a_frag_nc(Aprev, lda, mt * 16, k);
      v16bf br = load_bt_frag(Whh, H_DIM, 0 * H_DIM + nt * 16, k);
      v16bf bz = load_bt_frag(Whh, H_DIM, 1 * H_DIM + nt * 16, k);
      v16bf bn = load_bt_frag(Whh, H_DIM, 2 * H_DIM + nt * 16, k);
      accr = wmma_bf16(a, br, accr);
      accz = wmma_bf16(a, bz, accz);
      accn = wmma_bf16(a, bn, accn);
    }
  }

  int lane  = threadIdx.x & 31;
  int col   = nt * 16 + (lane & 15);
  int rbase = mt * 16 + ((lane & 16) ? 8 : 0);
  float bhr = bhh[col], bhz = bhh[H_DIM + col], bhn = bhh[2 * H_DIM + col];
#pragma unroll
  for (int e = 0; e < 8; ++e) {
    int m = rbase + e;
    long o = ((long)m * T_DIM + t) * H_DIM + col;
    if (m < NNODES) {
      long xrow = ((long)m * T_DIM + t) * (3 * H_DIM);
      float xr = (float)xp[xrow + col];
      float xz = (float)xp[xrow + H_DIM + col];
      float xn = (float)xp[xrow + 2 * H_DIM + col];
      float hp = (t > 0) ? seqF[o - H_DIM] : 0.0f;  // (m,t-1,col)
      float r = 1.0f / (1.0f + __expf(-(xr + accr[e] + bhr)));
      float z = 1.0f / (1.0f + __expf(-(xz + accz[e] + bhz)));
      float n = tanhf(xn + r * (accn[e] + bhn));
      float hn = (1.0f - z) * n + z * hp;
      seqF[o]  = hn;
      seqBf[o] = (__bf16)hn;
    } else {
      seqBf[o] = (__bf16)0.0f;  // keep pad rows zero for next step's A loads
    }
  }
}

// ---------------------------------------------------------------------------
// Small elementwise / reduction kernels
// ---------------------------------------------------------------------------
__global__ void f2bf(const float* __restrict__ s, __bf16* __restrict__ d, long n) {
  long i = blockIdx.x * (long)blockDim.x + threadIdx.x;
  if (i < n) d[i] = (__bf16)s[i];
}

// Tiled transpose: seqT[b][d][t] = (bf16) seqF[b][t][d].
__global__ void transpose_seq(const float* __restrict__ seqF,
                              __bf16* __restrict__ seqT) {
  __shared__ float tile[32][33];
  int b  = blockIdx.z;
  int t0 = blockIdx.x * 32, d0 = blockIdx.y * 32;
  int tx = threadIdx.x & 31, ty = threadIdx.x >> 5;  // 32 x 8
#pragma unroll
  for (int i = ty; i < 32; i += 8)
    tile[i][tx] = seqF[((long)b * T_DIM + t0 + i) * H_DIM + d0 + tx];
  __syncthreads();
#pragma unroll
  for (int i = ty; i < 32; i += 8)
    seqT[((long)b * H_DIM + d0 + i) * T_DIM + t0 + tx] = (__bf16)tile[tx][i];
}

// Softmax over u (time) per (b,d), then weighted sum with seq -> v[b,d].
// wT layout: [b][d][u] (sequential in u per thread).
__global__ void att_pool_enc(const float* __restrict__ wT, const float* __restrict__ bt,
                             const float* __restrict__ seqF,
                             float* __restrict__ v, __bf16* __restrict__ vbf) {
  int idx = blockIdx.x * blockDim.x + threadIdx.x;  // b*H + d
  if (idx >= NNODES * H_DIM) return;
  int b = idx >> 10, d = idx & (H_DIM - 1);
  const float* wrow = wT + (long)idx * T_DIM;
  float mx = -1e30f;
#pragma unroll 4
  for (int u = 0; u < T_DIM; ++u) mx = fmaxf(mx, wrow[u] + bt[u]);
  float s = 0.0f, acc = 0.0f;
#pragma unroll 4
  for (int u = 0; u < T_DIM; ++u) {
    float e = __expf(wrow[u] + bt[u] - mx);
    s += e;
    acc += e * seqF[((long)b * T_DIM + u) * H_DIM + d];
  }
  float r = acc / s;
  v[idx]   = r;
  vbf[idx] = (__bf16)r;
}

// Pool 20 stocks -> category vector (tiny 20x20 attention over the group).
__global__ void att_pool_cat(const float* __restrict__ v, const float* __restrict__ W,
                             const float* __restrict__ b,
                             float* __restrict__ cat, __bf16* __restrict__ catbf) {
  int idx = blockIdx.x * blockDim.x + threadIdx.x;  // c*H + d
  if (idx >= NCAT * H_DIM) return;
  int c = idx >> 10, d = idx & (H_DIM - 1);
  float wv[NPER];
  float mx = -1e30f;
  for (int u = 0; u < NPER; ++u) {
    float s = b[u];
    for (int tt = 0; tt < NPER; ++tt)
      s += v[((c * NPER + tt) << 10) + d] * W[u * NPER + tt];
    wv[u] = s;
    mx = fmaxf(mx, s);
  }
  float den = 0.0f, acc = 0.0f;
  for (int u = 0; u < NPER; ++u) {
    float e = __expf(wv[u] - mx);
    den += e;
    acc += e * v[((c * NPER + u) << 10) + d];
  }
  float r = acc / den;
  cat[idx]   = r;
  catbf[idx] = (__bf16)r;
}

__global__ void gat_alpha(const float* __restrict__ h, const float* __restrict__ as,
                          const float* __restrict__ ad,
                          float* __restrict__ als, float* __restrict__ ald, int Nn) {
  int i = blockIdx.x * blockDim.x + threadIdx.x;
  if (i >= Nn) return;
  float s = 0.0f, d = 0.0f;
  for (int k = 0; k < H_DIM; ++k) {
    float x = h[(long)i * H_DIM + k];
    s += x * as[k];
    d += x * ad[k];
  }
  als[i] = s;
  ald[i] = d;
}

// GAT aggregate exploiting the fully-connected-within-group (+self-loop)
// structure: incoming neighbourhood of node i == its whole group.
__global__ void gat_aggregate(const float* __restrict__ h, const float* __restrict__ als,
                              const float* __restrict__ ald, const float* __restrict__ bias,
                              float* __restrict__ out, int group) {
  int i  = blockIdx.x;
  int g0 = (i / group) * group;
  __shared__ float p[32];
  if (threadIdx.x == 0) {
    float mx = -1e30f;
    for (int j = 0; j < group; ++j) {
      float e = als[g0 + j] + ald[i];
      e = (e > 0.0f) ? e : 0.2f * e;  // leaky_relu 0.2
      p[j] = e;
      mx = fmaxf(mx, e);
    }
    float s = 0.0f;
    for (int j = 0; j < group; ++j) { p[j] = __expf(p[j] - mx); s += p[j]; }
    for (int j = 0; j < group; ++j) p[j] /= s;
  }
  __syncthreads();
  for (int d = threadIdx.x; d < H_DIM; d += blockDim.x) {
    float a = 0.0f;
    for (int j = 0; j < group; ++j) a += p[j] * h[(long)(g0 + j) * H_DIM + d];
    out[(long)i * H_DIM + d] = a + bias[d];
  }
}

__global__ void assemble_fusion(const float* __restrict__ v, const float* __restrict__ catO,
                                const float* __restrict__ innO, __bf16* __restrict__ fin) {
  long idx = blockIdx.x * (long)blockDim.x + threadIdx.x;
  if (idx >= (long)NNODES * 3 * H_DIM) return;
  int i = (int)(idx / (3 * H_DIM)), d = (int)(idx % (3 * H_DIM));
  float x;
  if (d < H_DIM)          x = v[i * H_DIM + d];
  else if (d < 2 * H_DIM) x = catO[(i / NPER) * H_DIM + (d - H_DIM)];
  else                    x = innO[i * H_DIM + (d - 2 * H_DIM)];
  fin[idx] = (__bf16)x;
}

__global__ void heads(const float* __restrict__ f, const float* __restrict__ regW,
                      const float* __restrict__ regB, const float* __restrict__ clsW,
                      const float* __restrict__ clsB, float* __restrict__ out) {
  int i = blockIdx.x;
  __shared__ float sr[256], sc[256];
  float r = 0.0f, c = 0.0f;
  for (int d = threadIdx.x; d < H_DIM; d += blockDim.x) {
    float x = f[(long)i * H_DIM + d];
    r += x * regW[d];
    c += x * clsW[d];
  }
  sr[threadIdx.x] = r; sc[threadIdx.x] = c;
  __syncthreads();
  for (int s = blockDim.x / 2; s > 0; s >>= 1) {
    if ((int)threadIdx.x < s) {
      sr[threadIdx.x] += sr[threadIdx.x + s];
      sc[threadIdx.x] += sc[threadIdx.x + s];
    }
    __syncthreads();
  }
  if (threadIdx.x == 0) {
    out[i]          = sr[0] + regB[0];
    out[NNODES + i] = 1.0f / (1.0f + __expf(-(sc[0] + clsB[0])));
  }
}

// ---------------------------------------------------------------------------
extern "C" void kernel_launch(void* const* d_in, const int* in_sizes, int n_in,
                              void* d_out, int out_size, void* d_ws, size_t ws_size,
                              hipStream_t stream) {
  (void)in_sizes; (void)n_in; (void)out_size; (void)ws_size;

  const float* weekly  = (const float*)d_in[0];
  const float* gWih    = (const float*)d_in[1];
  const float* gWhh    = (const float*)d_in[2];
  const float* gbih    = (const float*)d_in[3];
  const float* gbhh    = (const float*)d_in[4];
  const float* encW    = (const float*)d_in[5];
  const float* encb    = (const float*)d_in[6];
  const float* poolW   = (const float*)d_in[7];
  const float* poolb   = (const float*)d_in[8];
  const float* innerW  = (const float*)d_in[9];
  const float* innerAs = (const float*)d_in[10];
  const float* innerAd = (const float*)d_in[11];
  const float* innerB  = (const float*)d_in[12];
  const float* catW    = (const float*)d_in[13];
  const float* catAs   = (const float*)d_in[14];
  const float* catAd   = (const float*)d_in[15];
  const float* catB    = (const float*)d_in[16];
  const float* fusW    = (const float*)d_in[17];
  const float* fusB    = (const float*)d_in[18];
  const float* regW    = (const float*)d_in[19];
  const float* regB    = (const float*)d_in[20];
  const float* clsW    = (const float*)d_in[21];
  const float* clsB    = (const float*)d_in[22];
  // d_in[23]/d_in[24] (edge lists) unused: graph structure is dense-per-group.
  float* out = (float*)d_out;

  char* base = (char*)d_ws;
  size_t off = 0;
  auto take = [&](size_t bytes) -> char* {
    off = (off + 255) & ~(size_t)255;
    char* p = base + off;
    off += bytes;
    return p;
  };

  __bf16* Xbf      = (__bf16*)take((size_t)NNODES * T_DIM * IN_D * 2);
  __bf16* WihBf    = (__bf16*)take((size_t)3 * H_DIM * IN_D * 2);
  __bf16* WhhBf    = (__bf16*)take((size_t)3 * H_DIM * H_DIM * 2);
  __bf16* WtBf     = (__bf16*)take((size_t)T_DIM * T_DIM * 2);
  __bf16* innerWBf = (__bf16*)take((size_t)H_DIM * H_DIM * 2);
  __bf16* catWBf   = (__bf16*)take((size_t)H_DIM * H_DIM * 2);
  __bf16* fusWBf   = (__bf16*)take((size_t)H_DIM * 3 * H_DIM * 2);
  __bf16* xp       = (__bf16*)take((size_t)NNODES * T_DIM * 3 * H_DIM * 2);  // reused
  float*  seqF     = (float*)take((size_t)NNODES * T_DIM * H_DIM * 4);
  __bf16* seqBf    = (__bf16*)take((size_t)NPAD * T_DIM * H_DIM * 2);   // padded
  __bf16* seqT     = (__bf16*)take((size_t)NNODES * H_DIM * T_DIM * 2); // [b][d][t]
  float*  vF       = (float*)take((size_t)NNODES * H_DIM * 4);
  __bf16* vBf      = (__bf16*)take((size_t)NNODES * H_DIM * 2);
  float*  hInner   = (float*)take((size_t)NNODES * H_DIM * 4);
  float*  innerOut = (float*)take((size_t)NNODES * H_DIM * 4);
  float*  catVec   = (float*)take((size_t)NCAT * H_DIM * 4);
  __bf16* catVecBf = (__bf16*)take((size_t)NCAT * H_DIM * 2);
  float*  hCat     = (float*)take((size_t)NCAT * H_DIM * 4);
  float*  catOut   = (float*)take((size_t)NCAT * H_DIM * 4);
  float*  alsI     = (float*)take(NNODES * 4);
  float*  aldI     = (float*)take(NNODES * 4);
  float*  alsC     = (float*)take(NCAT * 4);
  float*  aldC     = (float*)take(NCAT * 4);
  __bf16* finBf    = (__bf16*)take((size_t)NNODES * 3 * H_DIM * 2);
  float*  fF       = (float*)take((size_t)NNODES * H_DIM * 4);
  // xp (315 MB) is dead after the scan; its region hosts wAttT (210 MB f32).
  float*  wAttT    = (float*)xp;

  auto cvt = [&](const float* s, __bf16* d, long n) {
    f2bf<<<(int)((n + 255) / 256), 256, 0, stream>>>(s, d, n);
  };
  cvt(weekly, Xbf, (long)NNODES * T_DIM * IN_D);
  cvt(gWih, WihBf, 3L * H_DIM * IN_D);
  cvt(gWhh, WhhBf, 3L * H_DIM * H_DIM);
  cvt(encW, WtBf, (long)T_DIM * T_DIM);
  cvt(innerW, innerWBf, (long)H_DIM * H_DIM);
  cvt(catW, catWBf, (long)H_DIM * H_DIM);
  cvt(fusW, fusWBf, (long)H_DIM * 3 * H_DIM);

  // 1) xp = X @ Wih^T + bih   (51200 x 3072, bf16 out, unguarded: M%16==0)
  {
    int M = NNODES * T_DIM, N = 3 * H_DIM, K = IN_D;
    int tiles = (M / 16) * (N / 16);
    wmma_gemm<false, false, __bf16><<<dim3((tiles + 7) / 8, 1), 256, 0, stream>>>(
        Xbf, IN_D, 0, WihBf, IN_D, 0, xp, N, 0, M, N, K, gbih);
  }

  // 2) GRU scan: 512 dependent steps, each a (112x1024)x(1024x3072) WMMA GEMM.
  for (int t = 0; t < T_DIM; ++t)
    gru_step<<<56, 256, 0, stream>>>(seqBf, seqF, WhhBf, xp, gbhh, t);

  // 3) Encoder attention: transpose seq, then wAttT[b] = seqT[b] @ Wt^T
  transpose_seq<<<dim3(T_DIM / 32, H_DIM / 32, NNODES), 256, 0, stream>>>(seqF, seqT);
  {
    int M = H_DIM, N = T_DIM, K = T_DIM;  // per batch b
    int tiles = (M / 16) * (N / 16);      // 2048
    wmma_gemm<false, false, float><<<dim3((tiles + 7) / 8, NNODES), 256, 0, stream>>>(
        seqT, T_DIM, (long)H_DIM * T_DIM, WtBf, T_DIM, 0,
        wAttT, T_DIM, (long)H_DIM * T_DIM, M, N, K, nullptr);
  }
  att_pool_enc<<<(NNODES * H_DIM + 255) / 256, 256, 0, stream>>>(wAttT, encb, seqF,
                                                                 vF, vBf);

  // 4) Inner GAT (groups of 20, fully connected + self-loop)
  {
    int tiles = ((NNODES + 15) / 16) * (H_DIM / 16);
    wmma_gemm<false, true, float><<<dim3((tiles + 7) / 8, 1), 256, 0, stream>>>(
        vBf, H_DIM, 0, innerWBf, H_DIM, 0, hInner, H_DIM, 0,
        NNODES, H_DIM, H_DIM, nullptr);
  }
  gat_alpha<<<1, 128, 0, stream>>>(hInner, innerAs, innerAd, alsI, aldI, NNODES);
  gat_aggregate<<<NNODES, 256, 0, stream>>>(hInner, alsI, aldI, innerB, innerOut, NPER);

  // 5) Category pooling + category GAT (5 nodes, fully connected + self-loop)
  att_pool_cat<<<(NCAT * H_DIM + 255) / 256, 256, 0, stream>>>(vF, poolW, poolb,
                                                               catVec, catVecBf);
  {
    int tiles = ((NCAT + 15) / 16) * (H_DIM / 16);
    wmma_gemm<false, true, float><<<dim3((tiles + 7) / 8, 1), 256, 0, stream>>>(
        catVecBf, H_DIM, 0, catWBf, H_DIM, 0, hCat, H_DIM, 0,
        NCAT, H_DIM, H_DIM, nullptr);
  }
  gat_alpha<<<1, 128, 0, stream>>>(hCat, catAs, catAd, alsC, aldC, NCAT);
  gat_aggregate<<<NCAT, 256, 0, stream>>>(hCat, alsC, aldC, catB, catOut, NCAT);

  // 6) Fusion MLP + heads
  assemble_fusion<<<(NNODES * 3 * H_DIM + 255) / 256, 256, 0, stream>>>(vF, catOut,
                                                                        innerOut, finBf);
  {
    int tiles = ((NNODES + 15) / 16) * (H_DIM / 16);
    wmma_gemm<true, true, float><<<dim3((tiles + 7) / 8, 1), 256, 0, stream>>>(
        finBf, 3 * H_DIM, 0, fusWBf, 3 * H_DIM, 0, fF, H_DIM, 0,
        NNODES, H_DIM, 3 * H_DIM, fusB);
  }
  heads<<<NNODES, 256, 0, stream>>>(fF, regW, regB, clsW, clsB, out);
}
